// RNN_23776938950816
// MI455X (gfx1250) — compile-verified, matrix operand order checked by reading, not run on hardware
//
#include <hip/hip_runtime.h>
#include <hip/hip_bf16.h>
#include <stdint.h>

// ---------------------------------------------------------------------------
// LMU/HiPPO RNN scan for MI455X (gfx1250).
//   - persistent kernel: 4 workgroups x 256 threads (8 waves), each owns a
//     16-row batch tile; h/c state live in LDS across all 1024 steps.
//   - all GEMMs via v_wmma_f32_16x16x32_bf16 (f32 accumulate).
//   - A_stack / weights pre-converted to bf16 in WMMA B-fragment-linear
//     layout (prep kernels) so every B operand is 2 coalesced b128 loads.
//   - x_{t+1}, Bd_{t+1} prefetched with global_load_async_to_lds_b128
//     (ASYNCcnt) overlapped with compute; drained via s_wait_asynccnt.
// Workspace: 1024*16*8*512 + 32*28*512 + 8*16*512 bf16  (~129 MB).
// ---------------------------------------------------------------------------

typedef __bf16 bf16;
typedef __attribute__((ext_vector_type(16))) bf16  v16bf;
typedef __attribute__((ext_vector_type(8)))  bf16  v8bf;
typedef __attribute__((ext_vector_type(8)))  float v8f;

#define X_DIM   128
#define H_DIM   512
#define C_DIM   256
#define Y_DIM   128
#define SEQ_LEN 1024

#define XC_STRIDE 392   // 384 + 8 pad (bf16) -> lane stride 196 dwords -> no bank conflicts
#define H_STRIDE  520   // 512 + 8 pad (bf16)
#define C_STRIDE  260   // 256 + 4 pad (f32)

__device__ __forceinline__ v16bf make_frag(const bf16* p0, const bf16* p1) {
  union { v16bf v; v8bf h[2]; } u;
  u.h[0] = *(const v8bf*)p0;
  u.h[1] = *(const v8bf*)p1;
  return u.v;
}

__device__ __forceinline__ v8f wmma_bf16(v16bf a, v16bf b, v8f c) {
  return __builtin_amdgcn_wmma_f32_16x16x32_bf16(false, a, false, b, (short)0, c,
                                                 false, false);
}

// Async global -> LDS copy of 16 bytes per lane (tracked by ASYNCcnt).
// LDS aperture maps the low 32 bits of the flat address to the LDS offset.
__device__ __forceinline__ void async_b128(const void* g, void* l) {
  uint32_t lds = (uint32_t)(uintptr_t)l;
  uint64_t ga  = (uint64_t)(uintptr_t)g;
  asm volatile("global_load_async_to_lds_b128 %0, %1, off"
               :: "v"(lds), "v"(ga) : "memory");
}

__device__ __forceinline__ void wait_async0() {
  asm volatile("s_wait_asynccnt 0" ::: "memory");
}

// ---------------------------------------------------------------------------
// Prep 1: A_stack (1024,256,256) f32 -> bf16, B-fragment-linear:
//   dst[(((t*16 + ntile)*8 + kstep)*32 + lane)*16 + e]
//   with n = ntile*16 + (lane&15), half = lane>>4,
//        k = kstep*32 + half*8 + (e<8 ? e : e+8)
//   B operand of (c @ A_d^T) needs Bt[n][k] = A_d[n][k]  (row-major source).
// ---------------------------------------------------------------------------
__global__ __launch_bounds__(256) void prep_A(const float* __restrict__ A,
                                              bf16* __restrict__ dst) {
  size_t i = (size_t)blockIdx.x * 256 + threadIdx.x;   // < 67,108,864
  int e    = (int)(i & 15);
  int lane = (int)((i >> 4) & 31);
  int ks   = (int)((i >> 9) & 7);
  int nt   = (int)((i >> 12) & 15);
  int t    = (int)(i >> 16);
  int n    = nt * 16 + (lane & 15);
  int half = lane >> 4;
  int k    = ks * 32 + half * 8 + (e < 8 ? e : e + 8);
  dst[i] = (bf16)A[((size_t)t * 256 + n) * 256 + k];
}

// ---------------------------------------------------------------------------
// Prep 2: fused [W_xh ; W_hh] (K=896, N=512) and W_hy (K=512, N=128) -> bf16
// B-fragment-linear (same per-fragment mapping as above).
// ---------------------------------------------------------------------------
__global__ __launch_bounds__(256) void prep_W(const float* __restrict__ Wxh,
                                              const float* __restrict__ Whh,
                                              const float* __restrict__ Why,
                                              bf16* __restrict__ w2,
                                              bf16* __restrict__ wy) {
  int i = blockIdx.x * 256 + threadIdx.x;              // < 524,288
  if (i < 32 * 28 * 512) {
    int e = i & 15, lane = (i >> 4) & 31, q = i >> 9;
    int ks = q % 28, nt = q / 28;
    int n = nt * 16 + (lane & 15);
    int half = lane >> 4;
    int k = ks * 32 + half * 8 + (e < 8 ? e : e + 8);
    float v = (k < 384) ? Wxh[(size_t)k * 512 + n]
                        : Whh[(size_t)(k - 384) * 512 + n];
    w2[i] = (bf16)v;
  } else {
    int j = i - 32 * 28 * 512;                         // < 65,536
    int e = j & 15, lane = (j >> 4) & 31, q = j >> 9;
    int ks = q & 15, nt = q >> 4;
    int n = nt * 16 + (lane & 15);
    int half = lane >> 4;
    int k = ks * 32 + half * 8 + (e < 8 ? e : e + 8);
    wy[j] = (bf16)Why[(size_t)k * 128 + n];
  }
}

// ---------------------------------------------------------------------------
// Main persistent scan kernel: gridDim.x = 4 (batch tiles of 16 rows).
// ---------------------------------------------------------------------------
__global__ __launch_bounds__(256) void lmu_rnn(
    const float* __restrict__ x,      // (64,1024,128)
    const float* __restrict__ ie,     // (128,1)
    const float* __restrict__ he,     // (512,1)
    const float* __restrict__ me,     // (256,1)
    const float* __restrict__ bh_g,   // (1,512)
    const float* __restrict__ by_g,   // (1,128)
    const float* __restrict__ Bstack, // (1024,256,1)
    const bf16*  __restrict__ Afrag,  // fragment-linear A_stack (bf16)
    const bf16*  __restrict__ W2frag, // fragment-linear [Wxh;Whh]
    const bf16*  __restrict__ Wyfrag, // fragment-linear Why
    float* __restrict__ out)          // (64,1024,128)
{
  __shared__ bf16  xc[16][XC_STRIDE];   // [x_t | c] bf16 (A operand GEMM1/2)
  __shared__ bf16  hb[16][H_STRIDE];    // h bf16      (A operand GEMM3/4)
  __shared__ float cb[16][C_STRIDE];    // c master f32
  __shared__ float xs[16 * 128];        // x_t f32 staging (async dest)
  __shared__ float Bd[2][256];          // B_d double buffer (async dest)
  __shared__ float s_ie[128], s_he[512], s_me[256], s_bh[512], s_by[128];
  __shared__ float uvec[16];
  __shared__ float up[16][17];

  const int tid  = threadIdx.x;
  const int lane = tid & 31;
  const int w    = tid >> 5;        // wave 0..7
  const int half = lane >> 4;       // K-half select in fragments
  const int nl   = lane & 15;       // row (A) / column (B,C,D) within tile
  const int b0   = blockIdx.x * 16; // batch row base
  const int m    = tid >> 4;        // row for the u-reduction phase
  const int j    = tid & 15;        // column-chunk for the u-reduction phase

  // ---- one-time init -------------------------------------------------------
  for (int i = tid; i < 128; i += 256) { s_ie[i] = ie[i]; s_by[i] = by_g[i]; }
  for (int i = tid; i < 512; i += 256) { s_he[i] = he[i]; s_bh[i] = bh_g[i]; }
  for (int i = tid; i < 256; i += 256) s_me[i] = me[i];
  for (int i = tid; i < 16 * H_STRIDE;  i += 256) (&hb[0][0])[i] = (bf16)0.0f;
  for (int i = tid; i < 16 * XC_STRIDE; i += 256) (&xc[0][0])[i] = (bf16)0.0f;
  for (int i = tid; i < 16 * C_STRIDE;  i += 256) (&cb[0][0])[i] = 0.0f;
  __syncthreads();

  // preload x_0 and Bd_0 asynchronously
  for (int c = tid; c < 512; c += 256) {            // 512 chunks of 16B
    int mm = c >> 5, ko = (c & 31) * 4;
    async_b128(&x[((size_t)(b0 + mm) * SEQ_LEN + 0) * X_DIM + ko],
               &xs[mm * 128 + ko]);
  }
  if (tid < 64)
    async_b128(&Bstack[tid * 4], &Bd[0][tid * 4]);
  wait_async0();
  __syncthreads();

  // ---- the scan ------------------------------------------------------------
  for (int t = 0; t < SEQ_LEN; ++t) {
    const int cur = t & 1, nxt = cur ^ 1;

    // (a) x_t f32 -> xc[:,0:128] bf16; start u partial (x-term)
    float up_acc = 0.0f;
    {
      const float* xr = &xs[m * 128 + j * 8];
      bf16* dst = &xc[m][j * 8];
#pragma unroll
      for (int q = 0; q < 8; ++q) {
        float v = xr[q];
        dst[q] = (bf16)v;
        up_acc += v * s_ie[j * 8 + q];
      }
    }
    __syncthreads();   // xs fully consumed; xc x-columns published

    // (b) kick async prefetch of x_{t+1}, Bd_{t+1} (overlaps all compute)
    if (t + 1 < SEQ_LEN) {
      for (int c = tid; c < 512; c += 256) {
        int mm = c >> 5, ko = (c & 31) * 4;
        async_b128(&x[((size_t)(b0 + mm) * SEQ_LEN + (t + 1)) * X_DIM + ko],
                   &xs[mm * 128 + ko]);
      }
      if (tid < 64)
        async_b128(&Bstack[(size_t)(t + 1) * 256 + tid * 4], &Bd[nxt][tid * 4]);
    }

    // (c) u = x@ie + h@he + c@me  (16 values; 16-thread tree per row)
#pragma unroll 4
    for (int q = 0; q < 32; ++q) { int k = j * 32 + q; up_acc += (float)hb[m][k] * s_he[k]; }
#pragma unroll 4
    for (int q = 0; q < 16; ++q) { int k = j * 16 + q; up_acc += cb[m][k] * s_me[k]; }
    up[m][j] = up_acc;
    __syncthreads();
    if (tid < 16) {
      float s = 0.0f;
#pragma unroll
      for (int q = 0; q < 16; ++q) s += up[tid][q];
      uvec[tid] = s;
    }

    // (d) GEMM1: c(16x256) @ A_d^T(256x256); wave w owns N-tiles 2w, 2w+1
    v8f acc0 = {}, acc1 = {};
    {
      const bf16* Abase = Afrag + (size_t)t * 16 * 8 * 512;
#pragma unroll
      for (int ks = 0; ks < 8; ++ks) {
        const bf16* ap = &xc[nl][128 + ks * 32 + half * 8];
        v16bf a = make_frag(ap, ap + 16);
        const bf16* p0 = Abase + ((size_t)(2 * w)     * 8 + ks) * 512 + lane * 16;
        const bf16* p1 = Abase + ((size_t)(2 * w + 1) * 8 + ks) * 512 + lane * 16;
        acc0 = wmma_bf16(a, make_frag(p0, p0 + 8), acc0);
        acc1 = wmma_bf16(a, make_frag(p1, p1 + 8), acc1);
      }
    }
    __syncthreads();   // all reads of old c done; uvec ready

    // (e) c_next = acc + c + u*Bd ; write cb (f32) and xc c-columns (bf16)
    {
#pragma unroll
      for (int r = 0; r < 8; ++r) {
        int mm = 8 * half + r;
        int n0 = (2 * w) * 16 + nl, n1 = (2 * w + 1) * 16 + nl;
        float c0 = acc0[r] + cb[mm][n0] + uvec[mm] * Bd[cur][n0];
        float c1 = acc1[r] + cb[mm][n1] + uvec[mm] * Bd[cur][n1];
        cb[mm][n0] = c0;  cb[mm][n1] = c1;
        xc[mm][128 + n0] = (bf16)c0;
        xc[mm][128 + n1] = (bf16)c1;
      }
    }
    __syncthreads();

    // (f) GEMM2+3: [x|c](16x384)@Wxh + h(16x512)@Whh; wave w owns tiles 4w..4w+3
    v8f hacc[4] = {};
#pragma unroll
    for (int ks = 0; ks < 28; ++ks) {
      const bf16* ap = (ks < 12) ? &xc[nl][ks * 32 + half * 8]
                                 : &hb[nl][(ks - 12) * 32 + half * 8];
      v16bf a = make_frag(ap, ap + 16);
#pragma unroll
      for (int tt = 0; tt < 4; ++tt) {
        const bf16* bp = W2frag + ((size_t)(4 * w + tt) * 28 + ks) * 512 + lane * 16;
        hacc[tt] = wmma_bf16(a, make_frag(bp, bp + 8), hacc[tt]);
      }
    }
    __syncthreads();   // all reads of old h done

    // (g) h_next = tanh(acc + b_h) -> hb (bf16)
#pragma unroll
    for (int tt = 0; tt < 4; ++tt) {
#pragma unroll
      for (int r = 0; r < 8; ++r) {
        int mm = 8 * half + r;
        int n  = 64 * w + tt * 16 + nl;
        hb[mm][n] = (bf16)tanhf(hacc[tt][r] + s_bh[n]);
      }
    }
    __syncthreads();

    // (h) GEMM4: y = h(16x512) @ Why(512x128) + b_y ; wave w owns N-tile w
    v8f yacc = {};
#pragma unroll
    for (int ks = 0; ks < 16; ++ks) {
      const bf16* ap = &hb[nl][ks * 32 + half * 8];
      v16bf a = make_frag(ap, ap + 16);
      const bf16* bp = Wyfrag + ((size_t)w * 16 + ks) * 512 + lane * 16;
      yacc = wmma_bf16(a, make_frag(bp, bp + 8), yacc);
    }
    {
      int n = 16 * w + nl;
      float bias = s_by[n];
#pragma unroll
      for (int r = 0; r < 8; ++r) {
        int mm = 8 * half + r;
        out[((size_t)(b0 + mm) * SEQ_LEN + t) * Y_DIM + n] = yacc[r] + bias;
      }
    }

    // drain the t+1 prefetch before next iteration reads xs/Bd[nxt]
    wait_async0();
    __syncthreads();
  }
}

// ---------------------------------------------------------------------------
extern "C" void kernel_launch(void* const* d_in, const int* in_sizes, int n_in,
                              void* d_out, int out_size, void* d_ws, size_t ws_size,
                              hipStream_t stream) {
  const float* x   = (const float*)d_in[0];
  const float* ie  = (const float*)d_in[1];
  const float* he  = (const float*)d_in[2];
  const float* me  = (const float*)d_in[3];
  const float* Wxh = (const float*)d_in[4];
  const float* Whh = (const float*)d_in[5];
  const float* bh  = (const float*)d_in[6];
  const float* Why = (const float*)d_in[7];
  const float* by  = (const float*)d_in[8];
  const float* Ast = (const float*)d_in[9];
  const float* Bst = (const float*)d_in[10];

  bf16* wsA  = (bf16*)d_ws;
  bf16* wsW2 = wsA + (size_t)SEQ_LEN * 16 * 8 * 512;  // 67,108,864 elems
  bf16* wsWy = wsW2 + (size_t)32 * 28 * 512;          //    458,752 elems
  // total ws: ~129 MB (bf16 A_stack fragments + weights)

  prep_A<<<262144, 256, 0, stream>>>(Ast, wsA);
  prep_W<<<2048,   256, 0, stream>>>(Wxh, Whh, Why, wsW2, wsWy);
  lmu_rnn<<<4, 256, 0, stream>>>(x, ie, he, me, bh, by, Bst,
                                 wsA, wsW2, wsWy, (float*)d_out);
}
